// KATrainPythiaModel_8160437862771
// MI455X (gfx1250) — matrix operand order repaired and checked by reading, not compiled, not run on hardware
//
#include <hip/hip_runtime.h>
#include <hip/hip_bf16.h>
#include <math.h>

typedef __attribute__((ext_vector_type(16))) _Float16 v16h;
typedef __attribute__((ext_vector_type(8)))  float    v8f;
typedef __attribute__((ext_vector_type(4)))  unsigned int u32x4;
typedef __attribute__((ext_vector_type(4)))  int      i32x4;
typedef __attribute__((ext_vector_type(8)))  int      i32x8;

#define L_  6
#define H_  512
#define NH_ 8
#define HD_ 64
#define FF_ 2048
#define V_  50304
#define B_  2
#define S_  1024
#define M_  (B_*S_)       // 2048 rows of activations
#define SCALE_ 0.125f     // HD^-0.5

// ---------------- f32 -> f16 conversion (once per operand; halves DRAM bytes)
__global__ void f32_to_f16_kernel(const float* __restrict__ src,
                                  _Float16* __restrict__ dst, long n) {
  long t = (long)blockIdx.x * 256 + threadIdx.x;
  if (t < n) dst[t] = (_Float16)src[t];
}

// ---------------- GEMM via WMMA (C = A * W^T + bias, optional GELU) ---------
// A: (M,K) f16 row-major, W: (N,K) f16 row-major, C: (M,N) fp32.
// Block tile 128x128, K-step 32, 8 waves (4 along M x 2 along N);
// each wave owns 32x64 -> 2x4 WMMA tiles = 8 v_wmma per K-step.
#define BM 128
#define BN 128
#define BK 32

union FragU { v16h h; unsigned int u[8]; };

// A fragment: 16x32 f16, lane l holds row m = l&15, K-half (l>>4)
__device__ inline v16h load_frag_A(const _Float16* As, int row0, int lane) {
  int m  = row0 + (lane & 15);
  int kh = (lane >> 4) * 16;
  const unsigned int* p = reinterpret_cast<const unsigned int*>(As + m * BK);
  FragU f;
#pragma unroll
  for (int v = 0; v < 8; ++v) {
    int kp = 2 * v + kh;   // even
    f.u[v] = p[kp >> 1];
  }
  return f.h;
}

// B fragment: 32x16 f16 (K x N), lane l holds col n = l&15, K-half (l>>4)*16
__device__ inline v16h load_frag_B(const _Float16* Bs, int col0, int lane) {
  int n  = col0 + (lane & 15);
  int kb = (lane >> 4) * 16;
  const unsigned int* p = reinterpret_cast<const unsigned int*>(Bs + n * BK);
  FragU f;
#pragma unroll
  for (int v = 0; v < 8; ++v) f.u[v] = p[(kb >> 1) + v];
  return f.h;
}

#if __has_builtin(__builtin_amdgcn_tensor_load_to_lds)
#define HAVE_TDM 1
// TDM DMA of a 2D f16 tile (rows x BK elements) from global to LDS.
// D# layout per CDNA5 ISA ch.8: group0 = count/lds/global/type,
// group1 = data_size, tensor dims, tile dims, dim0 stride.
__device__ inline void tdm_load_2d(unsigned lds_addr, const _Float16* gptr,
                                   int rows, long ld_elems,
                                   int tensor_d0, int tensor_d1) {
  unsigned long long ga = (unsigned long long)(uintptr_t)gptr;
  u32x4 g0;
  g0[0] = 1u;                                    // count=1, user descriptor
  g0[1] = lds_addr;                              // LDS byte address
  g0[2] = (unsigned)(ga & 0xffffffffull);        // global_addr[31:0]
  g0[3] = (unsigned)((ga >> 32) & 0x1ffffffull)  // global_addr[56:32]
        | (2u << 30);                            // type = 2 ("image")
  i32x8 g1;
  g1[0] = (int)(1u << 16);                       // data_size = 1 -> 2 bytes
  g1[1] = (int)(((unsigned)tensor_d0 & 0xffffu) << 16);          // dim0 lo16
  g1[2] = (int)((((unsigned)tensor_d0 >> 16) & 0xffffu)          // dim0 hi16
        |      (((unsigned)tensor_d1 & 0xffffu) << 16));         // dim1 lo16
  g1[3] = (int)((((unsigned)tensor_d1 >> 16) & 0xffffu)          // dim1 hi16
        |      ((unsigned)BK << 16));                            // tile_dim0
  g1[4] = (int)((unsigned)rows & 0xffffu);                       // tile_dim1
  g1[5] = (int)((unsigned long long)ld_elems & 0xffffffffull);   // stride lo
  g1[6] = (int)(((unsigned long long)ld_elems >> 32) & 0xffffull); // stride hi
  g1[7] = 0;
  i32x4 z4 = {0, 0, 0, 0};
#if __clang_major__ >= 23
  i32x8 z8 = {0, 0, 0, 0, 0, 0, 0, 0};
  __builtin_amdgcn_tensor_load_to_lds(g0, g1, z4, z4, z8, 0);
#else
  __builtin_amdgcn_tensor_load_to_lds(g0, g1, z4, z4, 0);
#endif
}
#else
#define HAVE_TDM 0
#endif

__global__ __launch_bounds__(256) void gemm_wmma16_kernel(
    const _Float16* __restrict__ A, long lda, long strideA,
    const _Float16* __restrict__ W, long ldw, long strideB,
    const float* __restrict__ bias,
    float* __restrict__ C, long ldc, long strideC,
    int M, int N, int K, int mode)   // mode: 0 = none, 1 = exact GELU
{
  __shared__ _Float16 As[BM * BK];   // 8 KB
  __shared__ _Float16 Bs[BN * BK];   // 8 KB

  long bz = blockIdx.z;
  A += bz * strideA;  W += bz * strideB;  C += bz * strideC;

  const int m0 = blockIdx.y * BM, n0 = blockIdx.x * BN;
  const int tid = threadIdx.x, lane = tid & 31, wave = tid >> 5;
  const int wm = (wave & 3) * 32;   // wave row offset within block tile
  const int wn = (wave >> 2) * 64;  // wave col offset

  v8f acc[2][4];
#pragma unroll
  for (int i = 0; i < 2; ++i)
#pragma unroll
    for (int j = 0; j < 4; ++j)
#pragma unroll
      for (int e = 0; e < 8; ++e) acc[i][j][e] = 0.0f;

#if HAVE_TDM
  unsigned ldsA = (unsigned)(uintptr_t)&As[0];
  unsigned ldsB = (unsigned)(uintptr_t)&Bs[0];
#endif

  for (int k0 = 0; k0 < K; k0 += BK) {
#if HAVE_TDM
    // wave 0 DMAs both tiles via the Tensor Data Mover, waits TENSORcnt,
    // then the workgroup barrier publishes the LDS tiles to all 8 waves.
    if (tid < 32) {
      tdm_load_2d(ldsA, A + (long)m0 * lda + k0, BM, lda, K, M);
      tdm_load_2d(ldsB, W + (long)n0 * ldw + k0, BN, ldw, K, N);
      __builtin_amdgcn_s_wait_tensorcnt(0);
    }
#else
    // fallback: cooperative staging, 16-byte vector copies of f16 rows
#pragma unroll
    for (int it = 0; it < 2; ++it) {
      int idx = tid + it * 256;
      int r = idx >> 2, c = (idx & 3) * 8;
      *(int4*)(As + r * BK + c) = *(const int4*)(A + (long)(m0 + r) * lda + k0 + c);
    }
#pragma unroll
    for (int it = 0; it < 2; ++it) {
      int idx = tid + it * 256;
      int r = idx >> 2, c = (idx & 3) * 8;
      *(int4*)(Bs + r * BK + c) = *(const int4*)(W + (long)(n0 + r) * ldw + k0 + c);
    }
#endif
    __syncthreads();

    v16h a0 = load_frag_A(As, wm,      lane);
    v16h a1 = load_frag_A(As, wm + 16, lane);
    v16h b0 = load_frag_B(Bs, wn,      lane);
    v16h b1 = load_frag_B(Bs, wn + 16, lane);
    v16h b2 = load_frag_B(Bs, wn + 32, lane);
    v16h b3 = load_frag_B(Bs, wn + 48, lane);

    acc[0][0] = __builtin_amdgcn_wmma_f32_16x16x32_f16(false, a0, false, b0, (short)0, acc[0][0], false, false);
    acc[0][1] = __builtin_amdgcn_wmma_f32_16x16x32_f16(false, a0, false, b1, (short)0, acc[0][1], false, false);
    acc[0][2] = __builtin_amdgcn_wmma_f32_16x16x32_f16(false, a0, false, b2, (short)0, acc[0][2], false, false);
    acc[0][3] = __builtin_amdgcn_wmma_f32_16x16x32_f16(false, a0, false, b3, (short)0, acc[0][3], false, false);
    acc[1][0] = __builtin_amdgcn_wmma_f32_16x16x32_f16(false, a1, false, b0, (short)0, acc[1][0], false, false);
    acc[1][1] = __builtin_amdgcn_wmma_f32_16x16x32_f16(false, a1, false, b1, (short)0, acc[1][1], false, false);
    acc[1][2] = __builtin_amdgcn_wmma_f32_16x16x32_f16(false, a1, false, b2, (short)0, acc[1][2], false, false);
    acc[1][3] = __builtin_amdgcn_wmma_f32_16x16x32_f16(false, a1, false, b3, (short)0, acc[1][3], false, false);
    __syncthreads();
  }

  // epilogue: D layout — VGPR v: lanes 0-15 -> M=v, lanes 16-31 -> M=8+v
#pragma unroll
  for (int mi = 0; mi < 2; ++mi)
#pragma unroll
    for (int ni = 0; ni < 4; ++ni) {
      int n = n0 + wn + ni * 16 + (lane & 15);
#pragma unroll
      for (int v = 0; v < 8; ++v) {
        int m = m0 + wm + mi * 16 + (lane >> 4) * 8 + v;
        float val = acc[mi][ni][v];
        if (bias) val += bias[n];
        if (mode == 1) val = 0.5f * val * (1.0f + erff(val * 0.70710678118f));
        C[(long)m * ldc + n] = val;
      }
    }
}

static void launch_gemm16(hipStream_t s,
                          const _Float16* A, long lda, long sA,
                          const _Float16* W, long ldw, long sB,
                          const float* bias,
                          float* C, long ldc, long sC,
                          int M, int N, int K, int mode, int batch) {
  dim3 grid(N / BN, M / BM, batch);
  gemm_wmma16_kernel<<<grid, 256, 0, s>>>(A, lda, sA, W, ldw, sB, bias, C, ldc, sC, M, N, K, mode);
}

static void convert_f16(hipStream_t s, const float* src, _Float16* dst, long n) {
  f32_to_f16_kernel<<<dim3((unsigned)((n + 255) / 256)), 256, 0, s>>>(src, dst, n);
}

// ---------------- small kernels --------------------------------------------
__global__ void embed_kernel(const int* __restrict__ ids,
                             const float* __restrict__ table,
                             float* __restrict__ h) {
  int row = blockIdx.x;                       // 0..M_-1
  long id = ids[row];
  const float* src = table + id * (long)H_;
  float* dst = h + (long)row * H_;
  for (int i = threadIdx.x; i < H_; i += blockDim.x) dst[i] = src[i];
}

__global__ __launch_bounds__(256) void ln_kernel(const float* __restrict__ x,
                                                 const float* __restrict__ g,
                                                 const float* __restrict__ b,
                                                 float* __restrict__ y, int Hd) {
  int row = blockIdx.x, t = threadIdx.x;
  const float* xr = x + (long)row * Hd;
  __shared__ float red[256];
  float s = 0.f;
  for (int i = t; i < Hd; i += 256) s += xr[i];
  red[t] = s; __syncthreads();
  for (int o = 128; o > 0; o >>= 1) { if (t < o) red[t] += red[t + o]; __syncthreads(); }
  float mean = red[0] / Hd;
  __syncthreads();
  float v = 0.f;
  for (int i = t; i < Hd; i += 256) { float d = xr[i] - mean; v += d * d; }
  red[t] = v; __syncthreads();
  for (int o = 128; o > 0; o >>= 1) { if (t < o) red[t] += red[t + o]; __syncthreads(); }
  float inv = rsqrtf(red[0] / Hd + 1e-5f);
  for (int i = t; i < Hd; i += 256)
    y[(long)row * Hd + i] = (xr[i] - mean) * inv * g[i] + b[i];
}

__global__ void rope_kernel(float* __restrict__ qkv) {
  int tt = blockIdx.x * 256 + threadIdx.x;    // M_*NH_*8 threads
  if (tt >= M_ * NH_ * 8) return;
  int j = tt & 7, head = (tt >> 3) & (NH_ - 1), m = tt >> 6;
  int s = m & (S_ - 1);
  float inv = powf(10000.f, -(2.f * j) / 16.f);
  float ang = (float)s * inv;
  float c = cosf(ang), sn = sinf(ang);
  float* q = qkv + (long)m * (3 * H_) + head * HD_;
  float* k = q + H_;
  float q1 = q[j], q2 = q[j + 8];
  q[j] = q1 * c - q2 * sn;  q[j + 8] = q2 * c + q1 * sn;
  float k1 = k[j], k2 = k[j + 8];
  k[j] = k1 * c - k2 * sn;  k[j + 8] = k2 * c + k1 * sn;
}

__global__ void split_heads_kernel(const float* __restrict__ qkv,
                                   float* __restrict__ qh,
                                   float* __restrict__ kh,
                                   float* __restrict__ vh) {
  long t = (long)blockIdx.x * 256 + threadIdx.x;   // M_*H_ elements
  if (t >= (long)M_ * H_) return;
  int d = t & (HD_ - 1);
  int head = (t >> 6) & (NH_ - 1);
  long m = t >> 9;
  int bb = (int)(m >> 10), s = (int)(m & (S_ - 1));
  long bh = (long)bb * NH_ + head;
  long dst = (bh * S_ + s) * HD_ + d;
  const float* src = qkv + m * (3 * H_) + head * HD_ + d;
  qh[dst] = src[0];
  kh[dst] = src[H_];
  vh[dst] = src[2 * H_];
}

__global__ __launch_bounds__(256) void softmax_causal_kernel(float* __restrict__ sc) {
  int r = blockIdx.x;            // bh*S + i
  int i = r & (S_ - 1);
  int t = threadIdx.x;
  float* row = sc + (long)r * S_;
  __shared__ float red[256];
  float mx = -3.4e38f;
  for (int j = t; j <= i; j += 256) mx = fmaxf(mx, row[j] * SCALE_);
  red[t] = mx; __syncthreads();
  for (int o = 128; o > 0; o >>= 1) { if (t < o) red[t] = fmaxf(red[t], red[t + o]); __syncthreads(); }
  float m = red[0];
  __syncthreads();
  float s = 0.f;
  for (int j = t; j <= i; j += 256) s += expf(row[j] * SCALE_ - m);
  red[t] = s; __syncthreads();
  for (int o = 128; o > 0; o >>= 1) { if (t < o) red[t] += red[t + o]; __syncthreads(); }
  float sm = red[0];
  for (int j = t; j < S_; j += 256)
    row[j] = (j <= i) ? expf(row[j] * SCALE_ - m) / sm : 0.f;
}

// sequential KA scan: one workgroup per (b,h); A state in global (L2-resident)
__global__ __launch_bounds__(256) void ka_scan_kernel(const float* __restrict__ w,
                                                      const float* __restrict__ vh,
                                                      float* __restrict__ A) {
  int bh = blockIdx.x;
  const float* wb = w  + (long)bh * S_ * S_;
  const float* vb = vh + (long)bh * S_ * HD_;
  float*       Ab = A  + (long)bh * S_ * HD_;
  __shared__ float wrow[S_];
  __shared__ float part[4 * HD_];
  __shared__ float red[256];
  int t = threadIdx.x, d = t & (HD_ - 1), g = t >> 6;

  if (t < HD_) Ab[t] = vb[t];            // A[0] = value[:, :, 0, :]
  __threadfence();
  __syncthreads();

  for (int i = 1; i < S_; ++i) {
    const float* wr = wb + (long)i * S_;
    float s = 0.f;
    for (int j = t; j < S_; j += 256) {
      float wv = (j < i) ? wr[j] : 0.f;  // strict causal mask for the scan
      wrow[j] = wv; s += wv;
    }
    red[t] = s; __syncthreads();
    for (int o = 128; o > 0; o >>= 1) { if (t < o) red[t] += red[t + o]; __syncthreads(); }
    float denom = red[0] + 1e-8f;

    float acc = 0.f;
    for (int j = g; j < i; j += 4) acc += wrow[j] * Ab[(long)j * HD_ + d];
    part[g * HD_ + d] = acc; __syncthreads();
    if (g == 0) {
      float tot = part[d] + part[HD_ + d] + part[2 * HD_ + d] + part[3 * HD_ + d];
      Ab[(long)i * HD_ + d] = tot / denom;
    }
    __threadfence();
    __syncthreads();
  }
}

__global__ void merge_heads_kernel(const float* __restrict__ A,
                                   float* __restrict__ ctx) {
  long t = (long)blockIdx.x * 256 + threadIdx.x;   // M_*H_ elements
  if (t >= (long)M_ * H_) return;
  int d = t & (HD_ - 1);
  int head = (t >> 6) & (NH_ - 1);
  long m = t >> 9;
  int bb = (int)(m >> 10), s = (int)(m & (S_ - 1));
  long bh = (long)bb * NH_ + head;
  ctx[m * H_ + head * HD_ + d] = A[(bh * S_ + s) * HD_ + d];
}

__global__ void add3_kernel(float* __restrict__ h,
                            const float* __restrict__ a,
                            const float* __restrict__ b) {
  long t = (long)blockIdx.x * 256 + threadIdx.x;
  if (t >= (long)M_ * H_) return;
  h[t] = h[t] + a[t] + b[t];
}

// ---------------- driver ----------------------------------------------------
extern "C" void kernel_launch(void* const* d_in, const int* in_sizes, int n_in,
                              void* d_out, int out_size, void* d_ws, size_t ws_size,
                              hipStream_t stream) {
  (void)in_sizes; (void)n_in; (void)out_size; (void)ws_size;
  const int*   ids     = (const int*)  d_in[0];
  const float* table   = (const float*)d_in[1];
  const float* qkv_w   = (const float*)d_in[2];
  const float* qkv_b   = (const float*)d_in[3];
  const float* dense_w = (const float*)d_in[4];
  const float* dense_b = (const float*)d_in[5];
  const float* fc1_w   = (const float*)d_in[6];
  const float* fc1_b   = (const float*)d_in[7];
  const float* fc2_w   = (const float*)d_in[8];
  const float* fc2_b   = (const float*)d_in[9];
  const float* ln_g    = (const float*)d_in[10];
  const float* ln_b    = (const float*)d_in[11];
  const float* fln_g   = (const float*)d_in[12];
  const float* fln_b   = (const float*)d_in[13];
  const float* out_w   = (const float*)d_in[14];
  float* logits = (float*)d_out;

  float* p = (float*)d_ws;
  float* h    = p; p += (size_t)M_ * H_;
  float* x    = p; p += (size_t)M_ * H_;
  float* qkv  = p; p += (size_t)M_ * 3 * H_;
  float* qh   = p; p += (size_t)B_ * NH_ * S_ * HD_;
  float* kh   = p; p += (size_t)B_ * NH_ * S_ * HD_;
  float* vh   = p; p += (size_t)B_ * NH_ * S_ * HD_;
  float* sc   = p; p += (size_t)B_ * NH_ * S_ * S_;
  float* Ab   = p; p += (size_t)B_ * NH_ * S_ * HD_;
  float* ctx  = p; p += (size_t)M_ * H_;
  float* aout = p; p += (size_t)M_ * H_;
  float* ff1  = p; p += (size_t)M_ * FF_;
  float* ff2  = p; p += (size_t)M_ * H_;
  _Float16* a16 = (_Float16*)p; p += ((size_t)M_ * FF_) / 2;   // max A: 2048x2048
  _Float16* w16 = (_Float16*)p; p += ((size_t)V_ * H_) / 2;    // max W: 50304x512

  embed_kernel<<<M_, 128, 0, stream>>>(ids, table, h);

  for (int l = 0; l < L_; ++l) {
    ln_kernel<<<M_, 256, 0, stream>>>(h, ln_g + l * H_, ln_b + l * H_, x, H_);

    // QKV projection
    convert_f16(stream, x, a16, (long)M_ * H_);
    convert_f16(stream, qkv_w + (size_t)l * 3 * H_ * H_, w16, (long)3 * H_ * H_);
    launch_gemm16(stream, a16, H_, 0, w16, H_, 0, qkv_b + (size_t)l * 3 * H_,
                  qkv, 3 * H_, 0, M_, 3 * H_, H_, 0, 1);

    rope_kernel<<<(M_ * NH_ * 8 + 255) / 256, 256, 0, stream>>>(qkv);
    split_heads_kernel<<<(M_ * H_ + 255) / 256, 256, 0, stream>>>(qkv, qh, kh, vh);

    // scores[bh] = q[bh] @ k[bh]^T, batched over 16 heads (scale in softmax)
    convert_f16(stream, qh, a16, (long)B_ * NH_ * S_ * HD_);
    convert_f16(stream, kh, w16, (long)B_ * NH_ * S_ * HD_);
    launch_gemm16(stream, a16, HD_, (long)S_ * HD_, w16, HD_, (long)S_ * HD_, nullptr,
                  sc, S_, (long)S_ * S_, S_, S_, HD_, 0, B_ * NH_);

    softmax_causal_kernel<<<B_ * NH_ * S_, 256, 0, stream>>>(sc);
    ka_scan_kernel<<<B_ * NH_, 256, 0, stream>>>(sc, vh, Ab);
    merge_heads_kernel<<<(M_ * H_ + 255) / 256, 256, 0, stream>>>(Ab, ctx);

    // attention output projection
    convert_f16(stream, ctx, a16, (long)M_ * H_);
    convert_f16(stream, dense_w + (size_t)l * H_ * H_, w16, (long)H_ * H_);
    launch_gemm16(stream, a16, H_, 0, w16, H_, 0, dense_b + (size_t)l * H_,
                  aout, H_, 0, M_, H_, H_, 0, 1);

    // MLP
    convert_f16(stream, x, a16, (long)M_ * H_);
    convert_f16(stream, fc1_w + (size_t)l * FF_ * H_, w16, (long)FF_ * H_);
    launch_gemm16(stream, a16, H_, 0, w16, H_, 0, fc1_b + (size_t)l * FF_,
                  ff1, FF_, 0, M_, FF_, H_, 1, 1);

    convert_f16(stream, ff1, a16, (long)M_ * FF_);
    convert_f16(stream, fc2_w + (size_t)l * H_ * FF_, w16, (long)H_ * FF_);
    launch_gemm16(stream, a16, FF_, 0, w16, FF_, 0, fc2_b + (size_t)l * H_,
                  ff2, H_, 0, M_, H_, FF_, 0, 1);

    add3_kernel<<<(M_ * H_) / 256, 256, 0, stream>>>(h, aout, ff2);
  }

  ln_kernel<<<M_, 256, 0, stream>>>(h, fln_g, fln_b, x, H_);
  convert_f16(stream, x, a16, (long)M_ * H_);
  convert_f16(stream, out_w, w16, (long)V_ * H_);
  launch_gemm16(stream, a16, H_, 0, w16, H_, 0, nullptr,
                logits, V_, 0, M_, V_, H_, 0, 1);
}
